// SelfAttentionLayer_19576460935837
// MI455X (gfx1250) — compile-verified, matrix operand order checked by reading, not run on hardware
//
#include <hip/hip_runtime.h>

typedef __attribute__((ext_vector_type(16))) _Float16 v16h;
typedef __attribute__((ext_vector_type(8)))  _Float16 v8h;
typedef __attribute__((ext_vector_type(8)))  float    v8f;

union F16x16 { v16h v; v8h h[2]; };

// Load a 16x32 f16 WMMA A-fragment (or B-fragment of a pre-transposed matrix)
// from row-major memory. Lanes 0-15: row=lane, K={0..7,16..23};
// lanes 16-31: row=lane&15, K={8..15,24..31}.
__device__ __forceinline__ v16h load_frag_g(const _Float16* __restrict__ p, int ld, int lane) {
  const int ln = lane & 15;
  const int ko = (lane >> 4) << 3;     // 0 or 8
  const _Float16* q = p + (size_t)ln * ld + ko;
  F16x16 f;
  f.h[0] = *(const v8h*)(q);
  f.h[1] = *(const v8h*)(q + 16);
  return f.v;
}

__device__ __forceinline__ v8f wmma_f16(v16h a, v16h b, v8f c) {
  return __builtin_amdgcn_wmma_f32_16x16x32_f16(false, a, false, b, (short)0, c, false, false);
}

// ---------------------------------------------------------------------------
__global__ __launch_bounds__(256) void cast_f32_f16(const float* __restrict__ in,
                                                    _Float16* __restrict__ out, int n) {
  const int i = blockIdx.x * 256 + threadIdx.x;
  if (i < n) out[i] = (_Float16)in[i];
}

// out[c][r] = (f16) in[r][c]; in is R x Ccols (fp32)
__global__ __launch_bounds__(256) void transpose_cast(const float* __restrict__ in,
                                                      _Float16* __restrict__ out,
                                                      int R, int Ccols) {
  __shared__ float tile[32][33];
  const int c0 = blockIdx.x * 32;
  const int r0 = blockIdx.y * 32;
  const int tx = threadIdx.x, ty = threadIdx.y;
#pragma unroll
  for (int i = 0; i < 4; ++i)
    tile[ty + i * 8][tx] = in[(size_t)(r0 + ty + i * 8) * Ccols + c0 + tx];
  __syncthreads();
#pragma unroll
  for (int i = 0; i < 4; ++i)
    out[(size_t)(c0 + ty + i * 8) * R + r0 + tx] = (_Float16)tile[tx][ty + i * 8];
}

// vT[b][c][t] = kqv[b*T+t][2048+c]   (f16)
__global__ __launch_bounds__(256) void transpose_v(const _Float16* __restrict__ kqv,
                                                   _Float16* __restrict__ vT) {
  __shared__ _Float16 tile[32][33];
  const int b = blockIdx.z;
  const int t0 = blockIdx.x * 32;
  const int c0 = blockIdx.y * 32;
  const int tx = threadIdx.x, ty = threadIdx.y;
#pragma unroll
  for (int i = 0; i < 4; ++i) {
    const int t = t0 + ty + i * 8;
    tile[ty + i * 8][tx] = kqv[((size_t)b * 2048 + t) * 3072 + 2048 + c0 + tx];
  }
  __syncthreads();
#pragma unroll
  for (int i = 0; i < 4; ++i) {
    const int c = c0 + ty + i * 8;
    vT[((size_t)b * 1024 + c) * 2048 + t0 + tx] = tile[tx][ty + i * 8];
  }
}

// ---------------------------------------------------------------------------
// WMMA GEMM: C[M x N] = A[M x K] * BT[N x K]^T + bias   (A, BT in f16)
// Block = 256 threads = 8 wave32 waves (4 row-groups x 2 col-groups),
// block tile 128 x 128, wave tile 32 x 64 (2x4 accumulators, 8 WMMA/stage).
// K-loop double-buffered: next stage's 6 fragments load under current WMMAs.
// ---------------------------------------------------------------------------
template <typename OutT>
__global__ __launch_bounds__(256) void gemm_wmma(const _Float16* __restrict__ A, int lda,
                                                 const _Float16* __restrict__ BT, int ldb,
                                                 const float* __restrict__ bias,
                                                 OutT* __restrict__ C, int N, int K) {
  const int tid = threadIdx.x;
  const int w = tid >> 5, lane = tid & 31;
  const int ln = lane & 15, hs = lane >> 4;
  const int rg = w >> 1, cg = w & 1;
  const int row0 = blockIdx.y * 128 + rg * 32;
  const int col0 = blockIdx.x * 128 + cg * 64;

  v8f acc[2][4];
#pragma unroll
  for (int i = 0; i < 2; ++i)
#pragma unroll
    for (int j = 0; j < 4; ++j)
#pragma unroll
      for (int e = 0; e < 8; ++e) acc[i][j][e] = 0.0f;

  v16h aF[2][2], bF[2][4];
#pragma unroll
  for (int i = 0; i < 2; ++i)
    aF[0][i] = load_frag_g(A + (size_t)(row0 + i * 16) * lda, lda, lane);
#pragma unroll
  for (int j = 0; j < 4; ++j)
    bF[0][j] = load_frag_g(BT + (size_t)(col0 + j * 16) * ldb, ldb, lane);

  for (int k = 0; k < K; k += 32) {
    const int cur = (k >> 5) & 1, nxt = cur ^ 1;
    const int k2 = k + 32;
    if (k2 < K) {  // prefetch next stage while current WMMAs run
#pragma unroll
      for (int i = 0; i < 2; ++i)
        aF[nxt][i] = load_frag_g(A + (size_t)(row0 + i * 16) * lda + k2, lda, lane);
#pragma unroll
      for (int j = 0; j < 4; ++j)
        bF[nxt][j] = load_frag_g(BT + (size_t)(col0 + j * 16) * ldb + k2, ldb, lane);
    }
#pragma unroll
    for (int i = 0; i < 2; ++i)
#pragma unroll
      for (int j = 0; j < 4; ++j)
        acc[i][j] = wmma_f16(aF[cur][i], bF[cur][j], acc[i][j]);
  }

#pragma unroll
  for (int j = 0; j < 4; ++j) {
    const int col = col0 + j * 16 + ln;
    const float bb = bias ? bias[col] : 0.0f;
#pragma unroll
    for (int i = 0; i < 2; ++i)
#pragma unroll
      for (int r = 0; r < 8; ++r) {
        const int row = row0 + i * 16 + r + 8 * hs;  // D layout: row = vgpr + 8*(lane>=16)
        C[(size_t)row * N + col] = (OutT)(acc[i][j][r] + bb);
      }
  }
}

// ---------------------------------------------------------------------------
// Flash attention (single head, d = C = 1024), causal, scale = 1/sqrt(T).
// Workgroup: one batch, 32 q rows. 8 waves = 2 row-groups x 4 col-groups.
// S-loop double-buffered; P*V loop uses 4-deep V-fragment ring (distance 3).
// ---------------------------------------------------------------------------
__global__ __launch_bounds__(256) void flash_attn(const _Float16* __restrict__ kqv, // [B*T,3072] k|q|v
                                                  const _Float16* __restrict__ vT,  // [B,1024,2048]
                                                  _Float16* __restrict__ outh) {    // [B*T,1024]
  constexpr int T = 2048, LD = 3072;
  const int b = blockIdx.y;
  const int q0 = blockIdx.x * 32;
  const int tid = threadIdx.x;
  const int w = tid >> 5, lane = tid & 31;
  const int ln = lane & 15, hs = lane >> 4;
  const int rg = w >> 2, cg = w & 3;

  const _Float16* qb = kqv + (size_t)b * T * LD + 1024;  // q slice
  const _Float16* kb = kqv + (size_t)b * T * LD;         // k slice
  const _Float16* vb = vT + (size_t)b * 1024 * T;

  __shared__ float smax[2][4][16];
  __shared__ float ssum[2][4][16];
  __shared__ __align__(16) _Float16 pl[32][72];  // P tile (144B rows, 16B aligned)

  v8f O[16];
#pragma unroll
  for (int n = 0; n < 16; ++n)
#pragma unroll
    for (int e = 0; e < 8; ++e) O[n][e] = 0.0f;

  float m[8], l[8];
#pragma unroll
  for (int r = 0; r < 8; ++r) { m[r] = -3.0e38f; l[r] = 0.0f; }

  const float scale = 0.02209708691207961f;  // 1/sqrt(2048)
  const int nkt = (q0 + 31) / 64 + 1;        // causal tile bound

  for (int kt = 0; kt < nkt; ++kt) {
    const int scol = kt * 64 + cg * 16;
    const _Float16* qrow = qb + (size_t)(q0 + rg * 16) * LD;
    const _Float16* krow = kb + (size_t)scol * LD;

    // ---- S = q * k^T for this wave's 16x16 tile, K = 1024 (double-buffered) ----
    v8f s;
#pragma unroll
    for (int e = 0; e < 8; ++e) s[e] = 0.0f;
    v16h aq[2], bk[2];
    aq[0] = load_frag_g(qrow, LD, lane);
    bk[0] = load_frag_g(krow, LD, lane);
    for (int c = 0; c < 1024; c += 32) {
      const int cur = (c >> 5) & 1, nxt = cur ^ 1;
      if (c + 32 < 1024) {
        aq[nxt] = load_frag_g(qrow + c + 32, LD, lane);
        bk[nxt] = load_frag_g(krow + c + 32, LD, lane);
      }
      s = wmma_f16(aq[cur], bk[cur], s);
    }

    // ---- scale + causal mask + per-row tile max (16-lane shuffle reduce) ----
    float tm[8];
#pragma unroll
    for (int r = 0; r < 8; ++r) {
      const int row = q0 + rg * 16 + r + 8 * hs;
      const int col = scol + ln;
      float v = s[r] * scale;
      if (col > row) v = -3.0e38f;
      s[r] = v;
      float x = v;
      for (int o = 8; o >= 1; o >>= 1) x = fmaxf(x, __shfl_xor(x, o, 16));
      tm[r] = x;
    }
    if (ln == 0) {
#pragma unroll
      for (int r = 0; r < 8; ++r) smax[rg][cg][r + 8 * hs] = tm[r];
    }
    __syncthreads();

    // ---- online softmax ----
    float mn[8], al[8], ts[8];
#pragma unroll
    for (int r = 0; r < 8; ++r) {
      const int rr = r + 8 * hs;
      float t = fmaxf(fmaxf(smax[rg][0][rr], smax[rg][1][rr]),
                      fmaxf(smax[rg][2][rr], smax[rg][3][rr]));
      mn[r] = fmaxf(m[r], t);
      al[r] = __expf(m[r] - mn[r]);
      const float p = __expf(s[r] - mn[r]);
      s[r] = p;
      float x = p;
      for (int o = 8; o >= 1; o >>= 1) x += __shfl_xor(x, o, 16);
      ts[r] = x;
      pl[rg * 16 + r + 8 * hs][cg * 16 + ln] = (_Float16)p;  // D-layout -> row-major stage
    }
    if (ln == 0) {
#pragma unroll
      for (int r = 0; r < 8; ++r) ssum[rg][cg][r + 8 * hs] = ts[r];
    }
    __syncthreads();

#pragma unroll
    for (int r = 0; r < 8; ++r) {
      const int rr = r + 8 * hs;
      const float t = ssum[rg][0][rr] + ssum[rg][1][rr] + ssum[rg][2][rr] + ssum[rg][3][rr];
      l[r] = l[r] * al[r] + t;
      m[r] = mn[r];
    }

    // ---- rescale O ----
#pragma unroll
    for (int n = 0; n < 16; ++n)
#pragma unroll
      for (int r = 0; r < 8; ++r) O[n][r] *= al[r];

    // ---- O += P * V : P A-frags from LDS up front, V B-frags in 4-deep ring ----
    v16h paF[2];
    {
      const int lrow = rg * 16 + ln;
#pragma unroll
      for (int ks = 0; ks < 2; ++ks) {
        const int c0 = ks * 32 + hs * 8;
        F16x16 t;
        t.h[0] = *(const v8h*)&pl[lrow][c0];      // ds_load_b128
        t.h[1] = *(const v8h*)&pl[lrow][c0 + 16];
        paF[ks] = t.v;
      }
    }
    const _Float16* vbase = vb + kt * 64;
    v16h vf[4];
#pragma unroll
    for (int j = 0; j < 3; ++j) {
      const int ks = j >> 4, n = j & 15;
      vf[j] = load_frag_g(vbase + (size_t)(cg * 256 + n * 16) * T + ks * 32, T, lane);
    }
#pragma unroll
    for (int idx = 0; idx < 32; ++idx) {
      const int pf = idx + 3;
      if (pf < 32) {
        const int ks = pf >> 4, n = pf & 15;
        vf[pf & 3] = load_frag_g(vbase + (size_t)(cg * 256 + n * 16) * T + ks * 32, T, lane);
      }
      O[idx & 15] = wmma_f16(paF[idx >> 4], vf[idx & 3], O[idx & 15]);
    }
    __syncthreads();  // protect pl/smax before next iteration rewrites them
  }

  // ---- epilogue: O / l -> f16 output ----
#pragma unroll
  for (int n = 0; n < 16; ++n) {
#pragma unroll
    for (int r = 0; r < 8; ++r) {
      const int row = b * T + q0 + rg * 16 + r + 8 * hs;
      const int col = cg * 256 + n * 16 + ln;
      outh[(size_t)row * 1024 + col] = (_Float16)(O[n][r] * (1.0f / l[r]));
    }
  }
}

// ---------------------------------------------------------------------------
extern "C" void kernel_launch(void* const* d_in, const int* in_sizes, int n_in,
                              void* d_out, int out_size, void* d_ws, size_t ws_size,
                              hipStream_t stream) {
  (void)in_sizes; (void)n_in; (void)out_size; (void)ws_size;
  const float* x      = (const float*)d_in[0];  // [4,2048,1024]
  const float* W_kqv  = (const float*)d_in[1];  // [1024,3072]
  const float* b_kqv  = (const float*)d_in[2];  // [3072]
  const float* W_proj = (const float*)d_in[3];  // [1024,1024]
  const float* b_proj = (const float*)d_in[4];  // [1024]
  float* out = (float*)d_out;                   // [4,2048,1024] fp32

  constexpr int Bb = 4, T = 2048, C = 1024;
  constexpr size_t M = (size_t)Bb * T;  // 8192

  char* ws = (char*)d_ws;
  _Float16* xh     = (_Float16*)ws;  ws += M * C * sizeof(_Float16);
  _Float16* WkqvT  = (_Float16*)ws;  ws += (size_t)3 * C * C * sizeof(_Float16);
  _Float16* WprojT = (_Float16*)ws;  ws += (size_t)C * C * sizeof(_Float16);
  _Float16* kqvh   = (_Float16*)ws;  ws += M * 3 * C * sizeof(_Float16);
  _Float16* vTt    = (_Float16*)ws;  ws += M * C * sizeof(_Float16);
  _Float16* attnh  = (_Float16*)ws;  ws += M * C * sizeof(_Float16);

  const int nx = (int)(M * C);
  cast_f32_f16<<<dim3((nx + 255) / 256), 256, 0, stream>>>(x, xh, nx);
  transpose_cast<<<dim3(3 * C / 32, C / 32), dim3(32, 8), 0, stream>>>(W_kqv, WkqvT, C, 3 * C);
  transpose_cast<<<dim3(C / 32, C / 32), dim3(32, 8), 0, stream>>>(W_proj, WprojT, C, C);

  gemm_wmma<_Float16><<<dim3(3 * C / 128, M / 128), 256, 0, stream>>>(
      xh, C, WkqvT, C, b_kqv, kqvh, 3 * C, C);

  transpose_v<<<dim3(T / 32, C / 32, Bb), dim3(32, 8), 0, stream>>>(kqvh, vTt);

  flash_attn<<<dim3(T / 32, Bb), 256, 0, stream>>>(kqvh, vTt, attnh);

  gemm_wmma<float><<<dim3(C / 128, M / 128), 256, 0, stream>>>(
      attnh, C, WprojT, C, b_proj, out, C, C);
}
